// EncoderBlock_31379031065248
// MI455X (gfx1250) — matrix operand order repaired and checked by reading, not compiled, not run on hardware
//
#include <hip/hip_runtime.h>
#include <hip/hip_bf16.h>

// ---------------------------------------------------------------------------
// Transformer encoder block for MI455X (gfx1250, wave32, WMMA).
// All GEMMs run through v_wmma_f32_16x16x32_f16 (f16 in, f32 accumulate).
// GEMM: 64x64 macro-tile per wave (16 WMMA per 16 b128 loads per K-step).
// ---------------------------------------------------------------------------

typedef __attribute__((ext_vector_type(16))) _Float16 v16h;
typedef __attribute__((ext_vector_type(8)))  _Float16 v8h;
typedef __attribute__((ext_vector_type(8)))  float    v8f;
typedef __attribute__((ext_vector_type(4)))  float    v4f_;

#define B_   2
#define S_   2048
#define D_   1024
#define H_   16
#define DK_  64
#define DFF_ 2048
#define M_   (B_ * S_)      // 4096 token rows
#define LSTR 2052           // padded LDS row stride (floats): 4-bank skew/row

// -------------------------- WMMA fragment helpers --------------------------
// A (16x32 f16, row-major source [*, K]):
//   lanes 0-15 : row m=lane,   elems 0..7 -> K=kb+0..7,  elems 8..15 -> K=kb+16..23
//   lanes16-31 : row m=lane-16,elems 0..7 -> K=kb+8..15, elems 8..15 -> K=kb+24..31
__device__ __forceinline__ v16h load_a_frag(const _Float16* __restrict__ A,
                                            int K, int m, int kb, int lane) {
  int lh = lane >> 4;
  const _Float16* p = A + (size_t)m * K + kb + 8 * lh;
  v8h lo = *(const v8h*)p;
  v8h hi = *(const v8h*)(p + 16);
  v16h a;
#pragma unroll
  for (int i = 0; i < 8; ++i) { a[i] = lo[i]; a[i + 8] = hi[i]; }
  return a;
}

// B (32x16 f16) from a row-major [N][K] (i.e. pre-transposed) source:
//   lane l holds column n = base + (l&15); elems 0..7 -> K = kb + 8*(l>>4) + e,
//   elems 8..15 -> K = kb + 16 + 8*(l>>4) + (e-8)
__device__ __forceinline__ v16h load_b_frag(const _Float16* __restrict__ Bt,
                                            int K, int n, int kb, int lane) {
  const _Float16* p = Bt + (size_t)n * K + kb + 8 * (lane >> 4);
  v8h lo = *(const v8h*)p;
  v8h hi = *(const v8h*)(p + 16);
  v16h b;
#pragma unroll
  for (int i = 0; i < 8; ++i) { b[i] = lo[i]; b[i + 8] = hi[i]; }
  return b;
}

// A-fragment sourced from fp32 scores in LDS (converted to f16 on the fly).
__device__ __forceinline__ v16h load_a_frag_lds(const float* __restrict__ sc,
                                                int m, int kb, int lane) {
  int lh = lane >> 4;
  const float* p = sc + m * LSTR + kb + 8 * lh;
  v4f_ l0 = *(const v4f_*)p;
  v4f_ l1 = *(const v4f_*)(p + 4);
  v4f_ h0 = *(const v4f_*)(p + 16);
  v4f_ h1 = *(const v4f_*)(p + 20);
  v16h a;
#pragma unroll
  for (int i = 0; i < 4; ++i) {
    a[i]      = (_Float16)l0[i];
    a[i + 4]  = (_Float16)l1[i];
    a[i + 8]  = (_Float16)h0[i];
    a[i + 12] = (_Float16)h1[i];
  }
  return a;
}

__device__ __forceinline__ v8f wmma_f16(v16h a, v16h b, v8f c) {
  return __builtin_amdgcn_wmma_f32_16x16x32_f16(false, a, false, b,
                                                (short)0, c, false, false);
}

// ------------------------ weight convert + transpose ------------------------
// Wt[n][k] = (f16) W[k][n]
__global__ void cvt_transpose_f16(const float* __restrict__ W,
                                  _Float16* __restrict__ Wt, int K, int N) {
  int idx = blockIdx.x * 256 + threadIdx.x;
  if (idx < K * N) {
    int k = idx / N, n = idx - k * N;
    Wt[(size_t)n * K + k] = (_Float16)W[idx];
  }
}

// ------------------------------- layernorm ---------------------------------
// one block per token row (D=1024, 256 threads x 4), f16 output for WMMA.
__global__ __launch_bounds__(256) void layernorm_f16(
    const float* __restrict__ x, const float* __restrict__ g,
    const float* __restrict__ bb, _Float16* __restrict__ y) {
  __shared__ float s1[256];
  __shared__ float s2[256];
  int tid = threadIdx.x;
  const float* xr = x + (size_t)blockIdx.x * D_;
  float v[4], sum = 0.f, sq = 0.f;
#pragma unroll
  for (int j = 0; j < 4; ++j) {
    v[j] = xr[tid + 256 * j];
    sum += v[j];
    sq  += v[j] * v[j];
  }
  s1[tid] = sum; s2[tid] = sq;
  __syncthreads();
  for (int st = 128; st > 0; st >>= 1) {
    if (tid < st) { s1[tid] += s1[tid + st]; s2[tid] += s2[tid + st]; }
    __syncthreads();
  }
  float mean = s1[0] * (1.f / D_);
  float var  = s2[0] * (1.f / D_) - mean * mean;
  float rs   = rsqrtf(var + 1e-5f);
  _Float16* yr = y + (size_t)blockIdx.x * D_;
#pragma unroll
  for (int j = 0; j < 4; ++j) {
    int c = tid + 256 * j;
    yr[c] = (_Float16)((v[j] - mean) * rs * g[c] + bb[c]);
  }
}

// ------------------------------ WMMA GEMM ----------------------------------
// C[M,N] = A[M,K] (f16, row-major) * Bt[N,K] (f16, pre-transposed) + bias.
// 4 waves/block in a 2x2 grid; each wave owns a 64x64 macro-tile (4x4 WMMA
// tiles -> 16 v_wmma per 16 global_load_b128 per 32-wide K step).
enum { MODE_QK = 0, MODE_V = 1, MODE_RES = 2, MODE_RELU = 3 };

template <int MODE>
__global__ __launch_bounds__(128) void gemm_wmma(
    const _Float16* __restrict__ A, const _Float16* __restrict__ Bt,
    const float* __restrict__ bias, const float* __restrict__ resid,
    float* __restrict__ outF, _Float16* __restrict__ outH,
    int M, int N, int K) {
  int tid  = threadIdx.x;
  int lane = tid & 31;
  int w    = tid >> 5;           // 0..3
  int wm   = w >> 1, wn = w & 1; // 2 (M) x 2 (N) wave grid
  int tileM = blockIdx.y * 128 + wm * 64;
  int tileN = blockIdx.x * 128 + wn * 64;
  int lm = lane & 15, lh = lane >> 4;

  v8f acc[4][4];
#pragma unroll
  for (int i = 0; i < 4; ++i)
#pragma unroll
    for (int j = 0; j < 4; ++j) acc[i][j] = (v8f){};

  for (int kb = 0; kb < K; kb += 32) {
    if (kb + 128 < K) {  // speculative prefetch of the K panel four steps out
      __builtin_prefetch(A + (size_t)(tileM + lm) * K + kb + 128, 0, 0);
      __builtin_prefetch(Bt + (size_t)(tileN + lm) * K + kb + 128, 0, 0);
    }
    v16h af[4], bf[4];
#pragma unroll
    for (int i = 0; i < 4; ++i)
      af[i] = load_a_frag(A, K, tileM + i * 16 + lm, kb, lane);
#pragma unroll
    for (int j = 0; j < 4; ++j)
      bf[j] = load_b_frag(Bt, K, tileN + j * 16 + lm, kb, lane);
#pragma unroll
    for (int i = 0; i < 4; ++i)
#pragma unroll
      for (int j = 0; j < 4; ++j)
        acc[i][j] = wmma_f16(af[i], bf[j], acc[i][j]);
  }

#pragma unroll
  for (int i = 0; i < 4; ++i) {
#pragma unroll
    for (int j = 0; j < 4; ++j) {
      int n = tileN + j * 16 + lm;
      float bn = bias[n];
#pragma unroll
      for (int e = 0; e < 8; ++e) {
        int m = tileM + i * 16 + e + 8 * lh;
        float val = acc[i][j][e] + bn;
        if (MODE == MODE_QK) {            // -> [b, h, s, dk]
          int b = m >> 11, q = m & 2047, h = n >> 6, dv = n & 63;
          outH[(((size_t)(b * H_ + h) * S_ + q) * DK_) + dv] = (_Float16)val;
        } else if (MODE == MODE_V) {      // -> [b, h, dk, s] (attn@V B-frag)
          int b = m >> 11, q = m & 2047, h = n >> 6, dv = n & 63;
          outH[(((size_t)(b * H_ + h) * DK_ + dv) * S_) + q] = (_Float16)val;
        } else if (MODE == MODE_RES) {    // fp32 out = residual + acc + bias
          size_t idx = (size_t)m * N + n;
          outF[idx] = resid[idx] + val;
        } else {                          // MODE_RELU: f16 out = relu(acc+bias)
          outH[(size_t)m * N + n] = (_Float16)fmaxf(val, 0.f);
        }
      }
    }
  }
}

// ------------------------------- attention ---------------------------------
// One block (4 waves, 128 thr) per (b*h, 16-query tile).
// Full 16x2048 fp32 score strip lives in LDS (~132 KB of the 320 KB WGP LDS).
__global__ __launch_bounds__(128) void attn_kernel(
    const _Float16* __restrict__ Q,   // [b,h,s,dk]
    const _Float16* __restrict__ Km,  // [b,h,s,dk]
    const _Float16* __restrict__ Vt,  // [b,h,dk,s]
    _Float16* __restrict__ Obug) {    // [b*s, d] with reference's swap/reshape
  extern __shared__ float smem[];
  float* sc  = smem;                  // 16 x LSTR
  float* red = smem + 16 * LSTR;      // 128
  float* inv = red + 128;             // 16

  int tid = threadIdx.x;
  int lane = tid & 31;
  int w = tid >> 5;                   // 0..3
  int qt = blockIdx.x;                // 0..127
  int bh = blockIdx.y;                // 0..31
  int b = bh >> 4, h = bh & 15;
  const _Float16* Qb = Q  + (size_t)bh * S_ * DK_;
  const _Float16* Kb = Km + (size_t)bh * S_ * DK_;
  const _Float16* Vb = Vt + (size_t)bh * DK_ * S_;
  int lm = lane & 15, lh = lane >> 4;
  int qrow = qt * 16 + lm;

  // ---- phase 1: scores = Q K^T / sqrt(dk), striped over the 4 waves ----
  v16h aq0 = load_a_frag(Qb, DK_, qrow, 0, lane);
  v16h aq1 = load_a_frag(Qb, DK_, qrow, 32, lane);
  for (int kt = w; kt < S_ / 16; kt += 4) {
    int kcol = kt * 16 + lm;
    v8f c0 = {}, c1 = {};
    c0 = wmma_f16(aq0, load_b_frag(Kb, DK_, kcol, 0, lane), c0);
    c1 = wmma_f16(aq1, load_b_frag(Kb, DK_, kcol, 32, lane), c1);
#pragma unroll
    for (int j = 0; j < 8; ++j)
      sc[(j + 8 * lh) * LSTR + kt * 16 + lm] = (c0[j] + c1[j]) * 0.125f;
  }
  __syncthreads();

  // ---- phase 2: softmax per row (keep exp unnormalized, fold 1/sum later) --
  int r = tid >> 3, t8 = tid & 7;     // 8 threads per query row
  float* row = sc + r * LSTR;
  float lmax = -1e30f;
  for (int ci = t8; ci < S_; ci += 8) lmax = fmaxf(lmax, row[ci]);
  red[tid] = lmax;
  __syncthreads();
  float rmax = red[r * 8];
#pragma unroll
  for (int i = 1; i < 8; ++i) rmax = fmaxf(rmax, red[r * 8 + i]);
  __syncthreads();
  float lsum = 0.f;
  for (int ci = t8; ci < S_; ci += 8) {
    float e = __expf(row[ci] - rmax);
    row[ci] = e;
    lsum += e;
  }
  red[tid] = lsum;
  __syncthreads();
  if (t8 == 0) {
    float s = 0.f;
#pragma unroll
    for (int i = 0; i < 8; ++i) s += red[r * 8 + i];
    inv[r] = 1.f / s;
  }
  __syncthreads();

  // ---- phase 3: O = attn @ V ; wave w owns dk columns w*16..w*16+15 ----
  int dn = w * 16 + lm;
  v8f c0 = {}, c1 = {};
  for (int kb = 0; kb < S_; kb += 64) {
    c0 = wmma_f16(load_a_frag_lds(sc, lm, kb, lane),
                  load_b_frag(Vb, S_, dn, kb, lane), c0);
    c1 = wmma_f16(load_a_frag_lds(sc, lm, kb + 32, lane),
                  load_b_frag(Vb, S_, dn, kb + 32, lane), c1);
  }
  // epilogue reproduces reference's swapaxes(-1,-2).reshape(b,s,d) mapping
#pragma unroll
  for (int j = 0; j < 8; ++j) {
    int m = j + 8 * lh;
    float val = (c0[j] + c1[j]) * inv[m];
    int qg = qt * 16 + m;
    size_t flat = (size_t)h * DK_ * S_ + (size_t)dn * S_ + qg;
    int irow = (int)(flat >> 10);
    int jcol = (int)(flat & 1023);
    Obug[((size_t)b * S_ + irow) * D_ + jcol] = (_Float16)val;
  }
}

// -------------------------------- launcher ---------------------------------
extern "C" void kernel_launch(void* const* d_in, const int* in_sizes, int n_in,
                              void* d_out, int out_size, void* d_ws,
                              size_t ws_size, hipStream_t stream) {
  const float* x   = (const float*)d_in[0];
  const float* Wq  = (const float*)d_in[1];
  const float* bq  = (const float*)d_in[2];
  const float* Wk  = (const float*)d_in[3];
  const float* bk  = (const float*)d_in[4];
  const float* Wv  = (const float*)d_in[5];
  const float* bv  = (const float*)d_in[6];
  const float* Wo  = (const float*)d_in[7];
  const float* bo  = (const float*)d_in[8];
  const float* g1  = (const float*)d_in[9];
  const float* be1 = (const float*)d_in[10];
  const float* g2  = (const float*)d_in[11];
  const float* be2 = (const float*)d_in[12];
  const float* W1  = (const float*)d_in[13];
  const float* b1  = (const float*)d_in[14];
  const float* W2  = (const float*)d_in[15];
  const float* b2  = (const float*)d_in[16];
  float* out = (float*)d_out;

  char* p = (char*)d_ws;
  auto take = [&](size_t bytes) -> void* {
    void* r = (void*)p;
    p += (bytes + 255) & ~(size_t)255;
    return r;
  };
  _Float16* y16 = (_Float16*)take((size_t)M_ * D_ * 2);     // LN out (reused for LN2)
  _Float16* WqT = (_Float16*)take((size_t)D_ * D_ * 2);
  _Float16* WkT = (_Float16*)take((size_t)D_ * D_ * 2);
  _Float16* WvT = (_Float16*)take((size_t)D_ * D_ * 2);
  _Float16* WoT = (_Float16*)take((size_t)D_ * D_ * 2);
  _Float16* W1T = (_Float16*)take((size_t)D_ * DFF_ * 2);
  _Float16* W2T = (_Float16*)take((size_t)DFF_ * D_ * 2);
  _Float16* Q16 = (_Float16*)take((size_t)M_ * D_ * 2);
  _Float16* K16 = (_Float16*)take((size_t)M_ * D_ * 2);     // contiguous after Q16
  _Float16* V16 = (_Float16*)take((size_t)M_ * D_ * 2);
  _Float16* O16 = (_Float16*)take((size_t)M_ * D_ * 2);
  float*    x1  = (float*)   take((size_t)M_ * D_ * 4);
  _Float16* h16 = Q16;   // FFN hidden (M_*DFF_ f16) reuses Q16+K16 region

  // weight convert + transpose (f32 [K][N] -> f16 [N][K])
  int t;
  t = D_ * D_;
  cvt_transpose_f16<<<(t + 255) / 256, 256, 0, stream>>>(Wq, WqT, D_, D_);
  cvt_transpose_f16<<<(t + 255) / 256, 256, 0, stream>>>(Wk, WkT, D_, D_);
  cvt_transpose_f16<<<(t + 255) / 256, 256, 0, stream>>>(Wv, WvT, D_, D_);
  cvt_transpose_f16<<<(t + 255) / 256, 256, 0, stream>>>(Wo, WoT, D_, D_);
  t = D_ * DFF_;
  cvt_transpose_f16<<<(t + 255) / 256, 256, 0, stream>>>(W1, W1T, D_, DFF_);
  t = DFF_ * D_;
  cvt_transpose_f16<<<(t + 255) / 256, 256, 0, stream>>>(W2, W2T, DFF_, D_);

  // sublayer 1: LN -> QKV -> attention -> Wo + residual
  layernorm_f16<<<M_, 256, 0, stream>>>(x, g1, be1, y16);
  dim3 gD(D_ / 128, M_ / 128);
  gemm_wmma<MODE_QK><<<gD, 128, 0, stream>>>(y16, WqT, bq, nullptr, nullptr,
                                             Q16, M_, D_, D_);
  gemm_wmma<MODE_QK><<<gD, 128, 0, stream>>>(y16, WkT, bk, nullptr, nullptr,
                                             K16, M_, D_, D_);
  gemm_wmma<MODE_V><<<gD, 128, 0, stream>>>(y16, WvT, bv, nullptr, nullptr,
                                            V16, M_, D_, D_);
  size_t shmem = (size_t)(16 * LSTR + 128 + 16) * sizeof(float);
  attn_kernel<<<dim3(S_ / 16, B_ * H_), 128, shmem, stream>>>(Q16, K16, V16, O16);
  gemm_wmma<MODE_RES><<<gD, 128, 0, stream>>>(O16, WoT, bo, x, x1, nullptr,
                                              M_, D_, D_);

  // sublayer 2: LN -> FFN -> residual
  layernorm_f16<<<M_, 256, 0, stream>>>(x1, g2, be2, y16);
  dim3 gF(DFF_ / 128, M_ / 128);
  gemm_wmma<MODE_RELU><<<gF, 128, 0, stream>>>(y16, W1T, b1, nullptr, nullptr,
                                               h16, M_, DFF_, D_);
  gemm_wmma<MODE_RES><<<gD, 128, 0, stream>>>(h16, W2T, b2, x1, out, nullptr,
                                              M_, D_, DFF_);
}
